// TokenMergingAttention_24215025614861
// MI455X (gfx1250) — matrix operand order repaired and checked by reading, not compiled, not run on hardware
//
#include <hip/hip_runtime.h>
#include <stdint.h>
#include <stddef.h>

// ---------------------------------------------------------------------------
// Problem constants (B=8, N=2048, C=768, heads=12, r=512, M=1536)
// ---------------------------------------------------------------------------
#define BB   8
#define NN   2048
#define CC   768
#define HH   12
#define DH   64
#define RR   512
#define MM   1536     // tokens after merge
#define NSRC 1024
#define NDST 1024

typedef __attribute__((ext_vector_type(16))) __bf16        bf16x16;
typedef __attribute__((ext_vector_type(8)))  float         f32x8;
typedef __attribute__((ext_vector_type(4)))  float         f32x4;
typedef __attribute__((ext_vector_type(4)))  unsigned int  u32x4;
typedef __attribute__((ext_vector_type(8)))  unsigned int  u32x8;
typedef __attribute__((ext_vector_type(2)))  unsigned int  u32x2;

__device__ __forceinline__ unsigned short f2bf(float f) {
    union { float f; unsigned int u; } v; v.f = f;
    unsigned int r = v.u + 0x7fffu + ((v.u >> 16) & 1u);   // RNE
    return (unsigned short)(r >> 16);
}

// ---------------------------------------------------------------------------
// CDNA5 async global->LDS copy (ASYNCcnt). 16B per enabled lane.
// LDS address = low 32 bits of the generic pointer (aperture in high bits).
// ---------------------------------------------------------------------------
__device__ __forceinline__ void async_copy_b128(void* lds, const void* gmem) {
    unsigned int       l = (unsigned int)(uintptr_t)lds;
    unsigned long long g = (unsigned long long)(uintptr_t)gmem;
    asm volatile("global_load_async_to_lds_b128 %0, %1, off"
                 :: "v"(l), "v"(g) : "memory");
}
__device__ __forceinline__ void async_wait0() {
    asm volatile("s_wait_asynccnt 0x0" ::: "memory");
}

__device__ __forceinline__ bf16x16 pack_frag(u32x4 lo, u32x4 hi) {
    u32x8 r;
    r[0]=lo[0]; r[1]=lo[1]; r[2]=lo[2]; r[3]=lo[3];
    r[4]=hi[0]; r[5]=hi[1]; r[6]=hi[2]; r[7]=hi[3];
    return __builtin_bit_cast(bf16x16, r);
}

// A-matrix 16x32 bf16 fragment: lanes 0-15 row M=lane, K={0..7,16..23};
// lanes 16-31 row M=lane-16, K={8..15,24..31}. src row-major, ld in elements.
__device__ __forceinline__ bf16x16 load_frag_a(const unsigned short* base, int ld) {
    int lane = threadIdx.x & 31;
    const unsigned short* p = base + (size_t)(lane & 15) * ld + ((lane >> 4) << 3);
    u32x4 lo = *(const u32x4*)(p);
    u32x4 hi = *(const u32x4*)(p + 16);
    return pack_frag(lo, hi);
}

// B-matrix 32x16 bf16 fragment: lane n<16 -> column n, K={0..7,16..23}; lane n+16 ->
// column n, K={8..15,24..31}. Fed with rows of B^T (row-major).
__device__ __forceinline__ bf16x16 load_frag_b(const unsigned short* base, int ld) {
    int lane = threadIdx.x & 31;
    const unsigned short* p = base + (size_t)(lane & 15) * ld + ((lane >> 4) << 3);
    u32x4 lo = *(const u32x4*)(p);
    u32x4 hi = *(const u32x4*)(p + 16);
    return pack_frag(lo, hi);
}

__device__ __forceinline__ f32x8 wmma_bf16(bf16x16 a, bf16x16 b, f32x8 c) {
    return __builtin_amdgcn_wmma_f32_16x16x32_bf16(false, a, false, b, (short)0, c,
                                                   false, false);
}

// ---------------------------------------------------------------------------
// 1) normalize metric rows, split even/odd, emit bf16
// ---------------------------------------------------------------------------
__global__ void prep_metric_kernel(const float* __restrict__ x,
                                   unsigned short* __restrict__ a_buf,
                                   unsigned short* __restrict__ b_buf) {
    int t = blockIdx.x;                  // 0..B*N-1
    int b = t / NN, n = t % NN;
    const float* row = x + (size_t)t * CC;
    float ss = 0.f;
    for (int c = threadIdx.x; c < CC; c += 256) { float v = row[c]; ss += v * v; }
    __shared__ float red[256];
    red[threadIdx.x] = ss;
    __syncthreads();
    for (int s = 128; s > 0; s >>= 1) {
        if (threadIdx.x < s) red[threadIdx.x] += red[threadIdx.x + s];
        __syncthreads();
    }
    float inv = rsqrtf(red[0] + 1e-24f);
    unsigned short* out = ((n & 1) ? b_buf : a_buf) + ((size_t)b * NSRC + (n >> 1)) * CC;
    for (int c = threadIdx.x; c < CC; c += 256) out[c] = f2bf(row[c] * inv);
}

// ---------------------------------------------------------------------------
// 2) weight transpose + bf16 convert: Wt[n][k] = W[k][n]
// ---------------------------------------------------------------------------
__global__ void transpose_bf16_kernel(const float* __restrict__ W,
                                      unsigned short* __restrict__ Wt,
                                      int K, int N) {
    size_t i = (size_t)blockIdx.x * 256 + threadIdx.x;
    if (i >= (size_t)K * N) return;
    int n = (int)(i / K), k = (int)(i % K);
    Wt[i] = f2bf(W[(size_t)k * N + n]);
}

// ---------------------------------------------------------------------------
// 3) scores = a_norm @ b_norm^T, fused running row-max/argmax.
//    128 rows x 64 cols per block; async double-buffered pipeline flattened
//    over (chunk, ktile). Fragment loads hoisted ahead of the WMMA burst.
// ---------------------------------------------------------------------------
__global__ void scores_argmax_kernel(const unsigned short* __restrict__ Ab,
                                     const unsigned short* __restrict__ Bb,
                                     float* __restrict__ node_max,
                                     int*   __restrict__ node_idx) {
    __shared__ __align__(16) unsigned short As[2][128 * 32];
    __shared__ __align__(16) unsigned short Bs[2][64 * 32];
    int tid = threadIdx.x, wave = tid >> 5, lane = tid & 31;
    int b = blockIdx.y;
    int m0 = blockIdx.x * 128;
    const unsigned short* Ap = Ab + (size_t)b * NSRC * CC;
    const unsigned short* Bp = Bb + (size_t)b * NDST * CC;

    int ra = tid >> 2, sa = (tid & 3) * 8;                 // A seg (u=0)
    int ra2 = (tid + 256) >> 2;                            // A seg (u=1)
    auto stage = [&](int buf, int n0s, int kts) {
        async_copy_b128(&As[buf][ra  * 32 + sa], Ap + (size_t)(m0 + ra ) * CC + kts + sa);
        async_copy_b128(&As[buf][ra2 * 32 + sa], Ap + (size_t)(m0 + ra2) * CC + kts + sa);
        async_copy_b128(&Bs[buf][ra  * 32 + sa], Bp + (size_t)(n0s + ra) * CC + kts + sa);
    };

    float rmax[8]; int ridx[8];
#pragma unroll
    for (int j = 0; j < 8; ++j) { rmax[j] = -1e30f; ridx[j] = 0; }

    f32x8 acc[4] = {};
    stage(0, 0, 0);
    int kt = 0, n0 = 0;
    const int TT = (NDST / 64) * (CC / 32);                // 384 tiles
    for (int tt = 0; tt < TT; ++tt) {
        int cur = tt & 1;
        int kt_n = kt + 32, n0_n = n0;
        if (kt_n == CC) { kt_n = 0; n0_n = n0 + 64; }
        async_wait0();
        __syncthreads();
        if (tt + 1 < TT) stage(cur ^ 1, n0_n, kt_n);
        if (kt == 0) {
#pragma unroll
            for (int fn = 0; fn < 4; ++fn) acc[fn] = f32x8{};
        }
        bf16x16 af = load_frag_a(&As[cur][wave * 16 * 32], 32);
        bf16x16 bfr[4];
#pragma unroll
        for (int fn = 0; fn < 4; ++fn)
            bfr[fn] = load_frag_b(&Bs[cur][fn * 16 * 32], 32);
#pragma unroll
        for (int fn = 0; fn < 4; ++fn)
            acc[fn] = wmma_bf16(af, bfr[fn], acc[fn]);
        if (kt == CC - 32) {
#pragma unroll
            for (int fn = 0; fn < 4; ++fn) {
                int col = n0 + fn * 16 + (lane & 15);
#pragma unroll
                for (int j = 0; j < 8; ++j) {
                    float v = acc[fn][j];
                    if (v > rmax[j]) { rmax[j] = v; ridx[j] = col; }
                }
            }
        }
        kt = kt_n; n0 = n0_n;
    }
    // reduce across the 16-lane group holding the same row
#pragma unroll
    for (int j = 0; j < 8; ++j) {
#pragma unroll
        for (int mk = 1; mk < 16; mk <<= 1) {
            float ov = __shfl_xor(rmax[j], mk);
            int   oi = __shfl_xor(ridx[j], mk);
            if (ov > rmax[j] || (ov == rmax[j] && oi < ridx[j])) { rmax[j] = ov; ridx[j] = oi; }
        }
    }
    if ((lane & 15) == 0) {
        int row = m0 + wave * 16 + ((lane >> 4) << 3);
#pragma unroll
        for (int j = 0; j < 8; ++j) {
            node_max[b * NSRC + row + j] = rmax[j];
            node_idx[b * NSRC + row + j] = ridx[j];
        }
    }
}

// ---------------------------------------------------------------------------
// 4) per-batch bitonic top-512 + mask / ranks / counts / src->y map
// ---------------------------------------------------------------------------
__global__ void topk_merge_kernel(const float* __restrict__ node_max,
                                  const int*   __restrict__ node_idx,
                                  int* __restrict__ src_idx,
                                  int* __restrict__ dst_idx,
                                  int* __restrict__ unm_idx,
                                  int* __restrict__ counts,
                                  int* __restrict__ srcmap) {
    __shared__ float vals[NSRC];
    __shared__ int   idxs[NSRC];
    __shared__ int   merged[NSRC];
    __shared__ int   dstc[NDST];
    __shared__ int   rankarr[NSRC];
    int b = blockIdx.x, tid = threadIdx.x;
    for (int p = tid; p < NSRC; p += 512) { vals[p] = node_max[b * NSRC + p]; idxs[p] = p; }
    __syncthreads();
    for (int k = 2; k <= NSRC; k <<= 1) {
        for (int jj = k >> 1; jj > 0; jj >>= 1) {
            for (int i = tid; i < NSRC; i += 512) {
                int ixj = i ^ jj;
                if (ixj > i) {
                    bool dir = ((i & k) == 0);        // descending blocks
                    float vi = vals[i], vj = vals[ixj];
                    int ii = idxs[i],  ij = idxs[ixj];
                    bool jGreater = (vj > vi) || (vj == vi && ij < ii);
                    if (jGreater == dir) {
                        vals[i] = vj; vals[ixj] = vi;
                        idxs[i] = ij; idxs[ixj] = ii;
                    }
                }
            }
            __syncthreads();
        }
    }
    for (int p = tid; p < NSRC; p += 512) { merged[p] = 0; dstc[p] = 1; }
    __syncthreads();
    if (tid < RR) {
        int s = idxs[tid];
        int d = node_idx[b * NSRC + s];
        src_idx[b * RR + tid] = s;
        dst_idx[b * RR + tid] = d;
        merged[s] = 1;
        atomicAdd(&dstc[d], 1);
    }
    __syncthreads();
    if (tid == 0) {
        int c = 0;
        for (int p = 0; p < NSRC; ++p) { rankarr[p] = c; if (!merged[p]) ++c; }
    }
    __syncthreads();
    for (int p = tid; p < NSRC; p += 512) {
        if (merged[p]) {
            srcmap[b * NSRC + p] = node_idx[b * NSRC + p];   // y row = matched dst
        } else {
            int rk = rankarr[p];
            srcmap[b * NSRC + p] = NDST + rk;                // y row = 1024 + rank
            unm_idx[b * RR + rk] = p;
        }
        counts[b * NDST + p] = dstc[p];
    }
}

// ---------------------------------------------------------------------------
// 5) merge: accum init (vectorized copy), atomic scatter, finalize -> xm bf16
// ---------------------------------------------------------------------------
__global__ void merge_init_kernel(const float* __restrict__ x, float* __restrict__ accum) {
    size_t i = (size_t)blockIdx.x * 256 + threadIdx.x;   // B*1024*192 float4s
    int c4 = (int)(i % (CC / 4)); size_t t = i / (CC / 4);
    int d = (int)(t % NDST); int b = (int)(t / NDST);
    ((f32x4*)accum)[i] =
        *(const f32x4*)(x + ((size_t)b * NN + 2 * d + 1) * CC + c4 * 4);
}

__global__ void merge_scatter_kernel(const float* __restrict__ x,
                                     const int* __restrict__ src_idx,
                                     const int* __restrict__ dst_idx,
                                     float* __restrict__ accum) {
    size_t i = (size_t)blockIdx.x * 256 + threadIdx.x;   // B*512*768
    int c = (int)(i % CC); size_t t = i / CC;
    int s = (int)(t % RR); int b = (int)(t / RR);
    int p = src_idx[b * RR + s];
    int d = dst_idx[b * RR + s];
    atomicAdd(&accum[((size_t)b * NDST + d) * CC + c],
              x[((size_t)b * NN + 2 * p) * CC + c]);
}

__global__ void merge_finalize_kernel(const float* __restrict__ x,
                                      const float* __restrict__ accum,
                                      const int* __restrict__ counts,
                                      const int* __restrict__ unm_idx,
                                      unsigned short* __restrict__ xm) {
    size_t i = (size_t)blockIdx.x * 256 + threadIdx.x;   // B*1536*192 groups of 4
    int c4 = (int)(i % (CC / 4)); size_t t = i / (CC / 4);
    int m = (int)(t % MM); int b = (int)(t / MM);
    f32x4 v;
    if (m < NDST) {
        v = *(const f32x4*)(accum + ((size_t)b * NDST + m) * CC + c4 * 4);
        float inv = 1.0f / (float)counts[b * NDST + m];
        v[0] *= inv; v[1] *= inv; v[2] *= inv; v[3] *= inv;
    } else {
        int p = unm_idx[b * RR + (m - NDST)];
        v = *(const f32x4*)(x + ((size_t)b * NN + 2 * p) * CC + c4 * 4);
    }
    u32x2 o;
    o[0] = (unsigned int)f2bf(v[0]) | ((unsigned int)f2bf(v[1]) << 16);
    o[1] = (unsigned int)f2bf(v[2]) | ((unsigned int)f2bf(v[3]) << 16);
    *(u32x2*)(xm + ((size_t)b * MM + m) * CC + c4 * 4) = o;
}

// ---------------------------------------------------------------------------
// 6) bf16 GEMM + bias: C = A(MxK) * Bt(NxK)^T. 128x128 block, 8 waves (4Mx2N),
//    each wave 32x64 via 2x4 WMMA frags. Async double-buffered LDS pipeline.
//    mode 0 -> f32 row-major; mode 1 -> scatter qkv (q/k row-major, v^T).
// ---------------------------------------------------------------------------
__global__ void gemm_bf16_kernel(const unsigned short* __restrict__ A,
                                 const unsigned short* __restrict__ Bt,
                                 const float* __restrict__ bias,
                                 float* __restrict__ Cf,
                                 unsigned short* __restrict__ Qo,
                                 unsigned short* __restrict__ Ko,
                                 unsigned short* __restrict__ Vt,
                                 int Ntot, int Kdim, int mode) {
    __shared__ __align__(16) unsigned short As[2][128 * 32];
    __shared__ __align__(16) unsigned short Bs[2][128 * 32];
    int tid = threadIdx.x, wave = tid >> 5, lane = tid & 31;
    int wm = wave >> 1, wn = wave & 1;
    int m0 = blockIdx.x * 128;
    int n0 = blockIdx.y * 128;

    int r0 = tid >> 2, s0 = (tid & 3) * 8;                 // seg u=0 (A), u=2 (B)
    int r1 = (tid + 256) >> 2;                             // seg u=1 (A), u=3 (B)
    auto stage = [&](int buf, int kts) {
        async_copy_b128(&As[buf][r0 * 32 + s0], A  + (size_t)(m0 + r0) * Kdim + kts + s0);
        async_copy_b128(&As[buf][r1 * 32 + s0], A  + (size_t)(m0 + r1) * Kdim + kts + s0);
        async_copy_b128(&Bs[buf][r0 * 32 + s0], Bt + (size_t)(n0 + r0) * Kdim + kts + s0);
        async_copy_b128(&Bs[buf][r1 * 32 + s0], Bt + (size_t)(n0 + r1) * Kdim + kts + s0);
    };

    f32x8 acc[2][4] = {};
    stage(0, 0);
    int nk = Kdim >> 5;
    for (int t = 0; t < nk; ++t) {
        int cur = t & 1;
        async_wait0();
        __syncthreads();
        if (t + 1 < nk) {
            __builtin_prefetch(A + (size_t)(m0 + r0) * Kdim + ((t + 1) << 5), 0, 0);
            stage(cur ^ 1, (t + 1) << 5);
        }
        bf16x16 a0 = load_frag_a(&As[cur][(wm * 32     ) * 32], 32);
        bf16x16 a1 = load_frag_a(&As[cur][(wm * 32 + 16) * 32], 32);
        bf16x16 bfr[4];
#pragma unroll
        for (int fn = 0; fn < 4; ++fn)
            bfr[fn] = load_frag_b(&Bs[cur][(wn * 64 + fn * 16) * 32], 32);
#pragma unroll
        for (int fn = 0; fn < 4; ++fn) {
            acc[0][fn] = wmma_bf16(a0, bfr[fn], acc[0][fn]);
            acc[1][fn] = wmma_bf16(a1, bfr[fn], acc[1][fn]);
        }
    }

#pragma unroll
    for (int fm = 0; fm < 2; ++fm) {
        int rbase = m0 + wm * 32 + fm * 16 + ((lane >> 4) << 3);
#pragma unroll
        for (int fn = 0; fn < 4; ++fn) {
            int c = n0 + wn * 64 + fn * 16 + (lane & 15);
            float bb = bias[c];
#pragma unroll
            for (int j = 0; j < 8; ++j) {
                float v = acc[fm][fn][j] + bb;
                int r = rbase + j;
                if (mode == 0) {
                    Cf[(size_t)r * Ntot + c] = v;
                } else {
                    int bidx = r / MM, t2 = r % MM;
                    int which = c / CC, ch = c % CC;
                    int hh = ch >> 6, d = ch & 63;
                    size_t bh = (size_t)bidx * HH + hh;
                    unsigned short bv = f2bf(v);
                    if (which == 0)      Qo[(bh * MM + t2) * DH + d] = bv;
                    else if (which == 1) Ko[(bh * MM + t2) * DH + d] = bv;
                    else                 Vt[(bh * DH + d) * MM + t2] = bv;
                }
            }
        }
    }
}

// ---------------------------------------------------------------------------
// 7) flash attention: per (b,h), 128 q-rows/block, 64-key tiles, online softmax.
//    QK^T and PV via WMMA; K/V tiles async double-buffered; P via per-wave LDS.
// ---------------------------------------------------------------------------
__global__ void flash_attn_kernel(const unsigned short* __restrict__ Qb,
                                  const unsigned short* __restrict__ Kb,
                                  const unsigned short* __restrict__ Vt,
                                  unsigned short* __restrict__ Ob) {
    __shared__ __align__(16) unsigned short Ks[2][64 * 64];
    __shared__ __align__(16) unsigned short Vs[2][64 * 64];
    __shared__ __align__(16) unsigned short Ps[8 * 16 * 64];
    int tid = threadIdx.x, wave = tid >> 5, lane = tid & 31;
    int bh = blockIdx.y;                 // b*12 + h
    int q0 = blockIdx.x * 128;
    int b = bh / HH, h = bh % HH;

    int rk0 = tid >> 3, sk0 = (tid & 7) * 8;               // K seg u=0, V seg u=2
    int rk1 = (tid + 256) >> 3;                            // K seg u=1, V seg u=3
    auto stage = [&](int buf, int kb) {
        async_copy_b128(&Ks[buf][rk0 * 64 + sk0],
                        Kb + ((size_t)bh * MM + kb + rk0) * DH + sk0);
        async_copy_b128(&Ks[buf][rk1 * 64 + sk0],
                        Kb + ((size_t)bh * MM + kb + rk1) * DH + sk0);
        async_copy_b128(&Vs[buf][rk0 * 64 + sk0],
                        Vt + ((size_t)bh * DH + rk0) * MM + kb + sk0);
        async_copy_b128(&Vs[buf][rk1 * 64 + sk0],
                        Vt + ((size_t)bh * DH + rk1) * MM + kb + sk0);
    };

    const unsigned short* qptr = Qb + ((size_t)bh * MM + q0 + wave * 16) * DH;
    bf16x16 qf0 = load_frag_a(qptr, DH);
    bf16x16 qf1 = load_frag_a(qptr + 32, DH);

    f32x8 Oa[4] = {};
    float m[8], l[8];
#pragma unroll
    for (int j = 0; j < 8; ++j) { m[j] = -1e30f; l[j] = 0.f; }

    stage(0, 0);
    const int NT = MM / 64;                                // 24 key tiles
    for (int t = 0; t < NT; ++t) {
        int cur = t & 1;
        async_wait0();
        __syncthreads();
        if (t + 1 < NT) stage(cur ^ 1, (t + 1) * 64);
        const unsigned short* ks = Ks[cur];
        const unsigned short* vs = Vs[cur];

        f32x8 sfr[4] = {};
        {
            bf16x16 kf[4];
#pragma unroll
            for (int fn = 0; fn < 4; ++fn) kf[fn] = load_frag_b(ks + fn * 16 * 64, 64);
#pragma unroll
            for (int fn = 0; fn < 4; ++fn) sfr[fn] = wmma_bf16(qf0, kf[fn], sfr[fn]);
#pragma unroll
            for (int fn = 0; fn < 4; ++fn) kf[fn] = load_frag_b(ks + fn * 16 * 64 + 32, 64);
#pragma unroll
            for (int fn = 0; fn < 4; ++fn) sfr[fn] = wmma_bf16(qf1, kf[fn], sfr[fn]);
        }

        float mnew[8];
#pragma unroll
        for (int j = 0; j < 8; ++j) {
            float v = m[j];
#pragma unroll
            for (int fn = 0; fn < 4; ++fn) v = fmaxf(v, sfr[fn][j] * 0.125f);
            v = fmaxf(v, __shfl_xor(v, 1));
            v = fmaxf(v, __shfl_xor(v, 2));
            v = fmaxf(v, __shfl_xor(v, 4));
            v = fmaxf(v, __shfl_xor(v, 8));
            mnew[j] = v;
        }

        unsigned short* pw = Ps + wave * 16 * 64;
        int prow = (lane >> 4) << 3;
        int pcol = lane & 15;
        float psum[8];
#pragma unroll
        for (int j = 0; j < 8; ++j) psum[j] = 0.f;
#pragma unroll
        for (int fn = 0; fn < 4; ++fn) {
#pragma unroll
            for (int j = 0; j < 8; ++j) {
                float p = __expf(sfr[fn][j] * 0.125f - mnew[j]);
                psum[j] += p;
                pw[(prow + j) * 64 + fn * 16 + pcol] = f2bf(p);
            }
        }
#pragma unroll
        for (int j = 0; j < 8; ++j) {
            float ps = psum[j];
            ps += __shfl_xor(ps, 1);
            ps += __shfl_xor(ps, 2);
            ps += __shfl_xor(ps, 4);
            ps += __shfl_xor(ps, 8);
            float alpha = __expf(m[j] - mnew[j]);
            l[j] = l[j] * alpha + ps;
            m[j] = mnew[j];
#pragma unroll
            for (int fd = 0; fd < 4; ++fd) Oa[fd][j] *= alpha;
        }

        bf16x16 pf0 = load_frag_a(pw, 64);
        bf16x16 pf1 = load_frag_a(pw + 32, 64);
        {
            bf16x16 vf[4];
#pragma unroll
            for (int fd = 0; fd < 4; ++fd) vf[fd] = load_frag_b(vs + fd * 16 * 64, 64);
#pragma unroll
            for (int fd = 0; fd < 4; ++fd) Oa[fd] = wmma_bf16(pf0, vf[fd], Oa[fd]);
#pragma unroll
            for (int fd = 0; fd < 4; ++fd) vf[fd] = load_frag_b(vs + fd * 16 * 64 + 32, 64);
#pragma unroll
            for (int fd = 0; fd < 4; ++fd) Oa[fd] = wmma_bf16(pf1, vf[fd], Oa[fd]);
        }
    }

    int rbase = q0 + wave * 16 + ((lane >> 4) << 3);
#pragma unroll
    for (int fd = 0; fd < 4; ++fd) {
        int d = fd * 16 + (lane & 15);
#pragma unroll
        for (int j = 0; j < 8; ++j) {
            float v = Oa[fd][j] / l[j];
            Ob[((size_t)b * MM + rbase + j) * CC + h * DH + d] = f2bf(v);
        }
    }
}

// ---------------------------------------------------------------------------
// 8) unmerge: gather y rows back into full (B, N, C) -- vectorized b128
// ---------------------------------------------------------------------------
__global__ void unmerge_kernel(const float* __restrict__ y,
                               const int* __restrict__ srcmap,
                               float* __restrict__ out) {
    size_t i = (size_t)blockIdx.x * 256 + threadIdx.x;   // B*2048*192 float4s
    int c4 = (int)(i % (CC / 4)); size_t t = i / (CC / 4);
    int n = (int)(t % NN); int b = (int)(t / NN);
    int yrow = (n & 1) ? (n >> 1) : srcmap[b * NSRC + (n >> 1)];
    ((f32x4*)out)[i] = *(const f32x4*)(y + ((size_t)b * MM + yrow) * CC + c4 * 4);
}

// ---------------------------------------------------------------------------
// launch
// ---------------------------------------------------------------------------
extern "C" void kernel_launch(void* const* d_in, const int* in_sizes, int n_in,
                              void* d_out, int out_size, void* d_ws, size_t ws_size,
                              hipStream_t stream) {
    (void)in_sizes; (void)n_in; (void)out_size; (void)ws_size;
    const float* x    = (const float*)d_in[0];
    const float* Wqkv = (const float*)d_in[1];
    const float* bqkv = (const float*)d_in[2];
    const float* Wo   = (const float*)d_in[3];
    const float* bo   = (const float*)d_in[4];
    float* out = (float*)d_out;

    char* ws = (char*)d_ws;
    size_t off = 0;
    auto carve = [&](size_t bytes) -> void* {
        void* p = ws + off;
        off = (off + bytes + 255) & ~(size_t)255;
        return p;
    };

    unsigned short* a_buf  = (unsigned short*)carve((size_t)BB * NSRC * CC * 2);
    unsigned short* b_buf  = (unsigned short*)carve((size_t)BB * NDST * CC * 2);
    float*          nmax   = (float*)carve((size_t)BB * NSRC * 4);
    int*            nidx   = (int*)carve((size_t)BB * NSRC * 4);
    int*            sidx   = (int*)carve((size_t)BB * RR * 4);
    int*            didx   = (int*)carve((size_t)BB * RR * 4);
    int*            uidx   = (int*)carve((size_t)BB * RR * 4);
    int*            cnts   = (int*)carve((size_t)BB * NDST * 4);
    int*            smap   = (int*)carve((size_t)BB * NSRC * 4);
    float*          accum  = (float*)carve((size_t)BB * NDST * CC * 4);
    unsigned short* xm     = (unsigned short*)carve((size_t)BB * MM * CC * 2);
    unsigned short* WqkvT  = (unsigned short*)carve((size_t)3 * CC * CC * 2);
    unsigned short* WoT    = (unsigned short*)carve((size_t)CC * CC * 2);
    unsigned short* qb     = (unsigned short*)carve((size_t)BB * HH * MM * DH * 2);
    unsigned short* kb     = (unsigned short*)carve((size_t)BB * HH * MM * DH * 2);
    unsigned short* vt     = (unsigned short*)carve((size_t)BB * HH * DH * MM * 2);
    unsigned short* ob     = (unsigned short*)carve((size_t)BB * MM * CC * 2);
    float*          yb     = (float*)carve((size_t)BB * MM * CC * 4);

    // 1) metric normalize -> bf16
    prep_metric_kernel<<<dim3(BB * NN), dim3(256), 0, stream>>>(x, a_buf, b_buf);

    // 2) weight transpose + convert
    transpose_bf16_kernel<<<dim3((3 * CC * CC + 255) / 256), dim3(256), 0, stream>>>(
        Wqkv, WqkvT, CC, 3 * CC);
    transpose_bf16_kernel<<<dim3((CC * CC + 255) / 256), dim3(256), 0, stream>>>(
        Wo, WoT, CC, CC);

    // 3) cosine scores + row argmax (WMMA, async double-buffer)
    scores_argmax_kernel<<<dim3(NSRC / 128, BB), dim3(256), 0, stream>>>(
        a_buf, b_buf, nmax, nidx);

    // 4) top-512 + mask/rank/count maps
    topk_merge_kernel<<<dim3(BB), dim3(512), 0, stream>>>(
        nmax, nidx, sidx, didx, uidx, cnts, smap);

    // 5) merge
    merge_init_kernel<<<dim3((BB * NDST * (CC / 4)) / 256), dim3(256), 0, stream>>>(x, accum);
    merge_scatter_kernel<<<dim3((BB * RR * CC) / 256), dim3(256), 0, stream>>>(
        x, sidx, didx, accum);
    merge_finalize_kernel<<<dim3((BB * MM * (CC / 4)) / 256), dim3(256), 0, stream>>>(
        x, accum, cnts, uidx, xm);

    // 6) QKV GEMM (WMMA, async double-buffer) with q/k/vT scatter epilogue
    gemm_bf16_kernel<<<dim3((BB * MM) / 128, (3 * CC) / 128), dim3(256), 0, stream>>>(
        xm, WqkvT, bqkv, nullptr, qb, kb, vt, 3 * CC, CC, 1);

    // 7) flash attention (WMMA, async double-buffer)
    flash_attn_kernel<<<dim3(MM / 128, BB * HH), dim3(256), 0, stream>>>(qb, kb, vt, ob);

    // 8) output projection (WMMA, async double-buffer) -> f32
    gemm_bf16_kernel<<<dim3((BB * MM) / 128, CC / 128), dim3(256), 0, stream>>>(
        ob, WoT, bo, yb, nullptr, nullptr, nullptr, CC, CC, 0);

    // 9) unmerge gather
    unmerge_kernel<<<dim3((BB * NN * (CC / 4)) / 256), dim3(256), 0, stream>>>(yb, smap, out);
}
